// ChemicalEncoder_49160195670615
// MI455X (gfx1250) — compile-verified, matrix operand order checked by reading, not run on hardware
//
#include <hip/hip_runtime.h>
#include <hip/hip_bf16.h>

// ---------------- problem constants (fixed by reference) ----------------
#define N_ATOMS 100000
#define N_EDGES 200000
#define N_MOLS  4096
#define D_V 133
#define D_E 14
#define DH  512
#define BN_EPS 1e-5f

#define NT_TOTAL 32          // 512 / 16 N-tiles
#define KTI 5                // input GEMM: K = 147 -> padded 160 = 5*32
#define KI  (KTI*32)
#define KTH 16               // hidden GEMM: K = 512 = 16*32
#define KH  (KTH*32)
#define KTO 21               // output GEMM: K = 645 -> padded 672 = 21*32
#define KO  (KTO*32)

typedef __attribute__((ext_vector_type(16))) __bf16        v16bf;
typedef __attribute__((ext_vector_type(8)))  float         v8f;
typedef __attribute__((ext_vector_type(8)))  unsigned int  v8u;

__device__ __forceinline__ unsigned short f2bf(float f) {
  unsigned u = __builtin_bit_cast(unsigned, f);
  unsigned r = u + 0x7FFFu + ((u >> 16) & 1u);   // round-to-nearest-even
  return (unsigned short)(r >> 16);
}

// A-fragment mapping for 16x32 bf16 A (row r in tile, k' = k%32):
//   lane = r + 16*((k'>>3)&1), h = (k'&7) | ((k'&16)>>1)
// B-fragment mapping for 32x16 bf16 B (lane, element h):
//   n = n0 + (lane&15),  k' = (h&7) | ((h&8)<<1) | ((lane&16)>>1)

// -------- pack W (Krows x 512, f32 row-major) into WMMA-B per-lane layout ----
// P layout: [kt][nt][lane][h] of u16(bf16); rows >= Krows are zero padding.
__global__ __launch_bounds__(256) void pack_weight_kernel(
    const float* __restrict__ W, unsigned short* __restrict__ P,
    int Krows, int ktiles)
{
  int idx = blockIdx.x * 256 + threadIdx.x;
  int total = ktiles * NT_TOTAL * 32 * 16;
  if (idx >= total) return;
  int h    = idx & 15;
  int lane = (idx >> 4) & 31;
  int nt   = (idx >> 9) & 31;
  int kt   = idx >> 14;
  int n  = nt * 16 + (lane & 15);
  int kk = (h & 7) | ((h & 8) << 1) | ((lane & 16) >> 1);
  int k  = kt * 32 + kk;
  float v = (k < Krows) ? W[(long)k * DH + n] : 0.0f;
  P[idx] = f2bf(v);
}

// ---------------- GEMM 1: H0 = [V[src];E] @ W_i, H = relu(H0) ----------------
// 32 rows (2 M-tiles) per block; 8 waves x 4 N-tiles = full DH width.
__global__ __launch_bounds__(256) void gemm_input_kernel(
    const float* __restrict__ V, const float* __restrict__ E,
    const int* __restrict__ src, const unsigned short* __restrict__ Wp,
    float* __restrict__ H0, float* __restrict__ H)
{
  __shared__ unsigned short Apack[2 * KTI * 32 * 16];   // 10 KB
  __shared__ int s_src[32];
  const int m0 = blockIdx.x * 32;
  const int t  = threadIdx.x;
  if (t < 32) s_src[t] = src[m0 + t];
  __syncthreads();

  for (int idx = t; idx < 32 * KI; idx += 256) {
    int r = idx / KI, k = idx - r * KI;
    float v;
    if (k < D_V)            v = V[(long)s_src[r] * D_V + k];
    else if (k < D_V + D_E) v = E[(long)(m0 + r) * D_E + (k - D_V)];
    else                    v = 0.0f;
    int mt = r >> 4, rr = r & 15;
    int kt = k >> 5, kk = k & 31;
    int lane = rr + ((kk & 8) << 1);
    int h    = (kk & 7) | ((kk & 16) >> 1);
    Apack[((mt * KTI + kt) * 32 + lane) * 16 + h] = f2bf(v);
  }
  __syncthreads();

  const int lane = t & 31, wave = t >> 5;
  v8f acc[2][4] = {};
  for (int kt = 0; kt < KTI; ++kt) {
    v16bf a0 = __builtin_bit_cast(v16bf, *(const v8u*)&Apack[(kt * 32 + lane) * 16]);
    v16bf a1 = __builtin_bit_cast(v16bf, *(const v8u*)&Apack[(((KTI + kt) * 32) + lane) * 16]);
#pragma unroll
    for (int j = 0; j < 4; ++j) {
      int nt = wave * 4 + j;
      v16bf b = __builtin_bit_cast(v16bf,
          *(const v8u*)&Wp[((long)(kt * NT_TOTAL + nt) * 32 + lane) * 16]);
      acc[0][j] = __builtin_amdgcn_wmma_f32_16x16x32_bf16(
          false, a0, false, b, (short)0, acc[0][j], false, false);
      acc[1][j] = __builtin_amdgcn_wmma_f32_16x16x32_bf16(
          false, a1, false, b, (short)0, acc[1][j], false, false);
    }
  }
  const int col = lane & 15, rowoff = (lane >> 4) * 8;
#pragma unroll
  for (int mt = 0; mt < 2; ++mt) {
#pragma unroll
    for (int j = 0; j < 4; ++j) {
      int n = (wave * 4 + j) * 16 + col;
#pragma unroll
      for (int v = 0; v < 8; ++v) {
        long o = (long)(m0 + mt * 16 + rowoff + v) * DH + n;
        float x = acc[mt][j][v];
        H0[o] = x;
        H[o]  = x > 0.0f ? x : 0.0f;
      }
    }
  }
}

// ----- GEMM 2: Hn = relu(H0 + (Mn[src] - Hc[rev]) @ W_h), K = 512 -----------
__global__ __launch_bounds__(256) void gemm_update_kernel(
    const float* __restrict__ Mn, const float* __restrict__ Hc,
    const int* __restrict__ src, const unsigned short* __restrict__ Wp,
    const float* __restrict__ H0, float* __restrict__ Hn)
{
  __shared__ unsigned short Apack[2 * KTH * 32 * 16];   // 32 KB
  __shared__ int s_src[32];
  const int m0 = blockIdx.x * 32;
  const int t  = threadIdx.x;
  if (t < 32) s_src[t] = src[m0 + t];
  __syncthreads();

  // vectorized builder: float4 loads of Mn[src] and Hc[rev] rows
  const float4* Mn4 = (const float4*)Mn;
  const float4* Hc4 = (const float4*)Hc;
  for (int idx = t; idx < 32 * (KH / 4); idx += 256) {
    int r  = idx >> 7;            // row in 32-row tile
    int kv = idx & 127;           // float4 index within row
    int e  = m0 + r;
    float4 mv = Mn4[(long)s_src[r] * (DH / 4) + kv];
    float4 hv = Hc4[(long)(e ^ 1) * (DH / 4) + kv];
    float vals[4] = { mv.x - hv.x, mv.y - hv.y, mv.z - hv.z, mv.w - hv.w };
    int mt = r >> 4, rr = r & 15;
#pragma unroll
    for (int c = 0; c < 4; ++c) {
      int k  = kv * 4 + c;
      int kt = k >> 5, kk = k & 31;
      int lane = rr + ((kk & 8) << 1);
      int h    = (kk & 7) | ((kk & 16) >> 1);
      Apack[((mt * KTH + kt) * 32 + lane) * 16 + h] = f2bf(vals[c]);
    }
  }
  __syncthreads();

  const int lane = t & 31, wave = t >> 5;
  v8f acc[2][4] = {};
  for (int kt = 0; kt < KTH; ++kt) {
    v16bf a0 = __builtin_bit_cast(v16bf, *(const v8u*)&Apack[(kt * 32 + lane) * 16]);
    v16bf a1 = __builtin_bit_cast(v16bf, *(const v8u*)&Apack[(((KTH + kt) * 32) + lane) * 16]);
#pragma unroll
    for (int j = 0; j < 4; ++j) {
      int nt = wave * 4 + j;
      v16bf b = __builtin_bit_cast(v16bf,
          *(const v8u*)&Wp[((long)(kt * NT_TOTAL + nt) * 32 + lane) * 16]);
      acc[0][j] = __builtin_amdgcn_wmma_f32_16x16x32_bf16(
          false, a0, false, b, (short)0, acc[0][j], false, false);
      acc[1][j] = __builtin_amdgcn_wmma_f32_16x16x32_bf16(
          false, a1, false, b, (short)0, acc[1][j], false, false);
    }
  }
  const int col = lane & 15, rowoff = (lane >> 4) * 8;
#pragma unroll
  for (int mt = 0; mt < 2; ++mt) {
#pragma unroll
    for (int j = 0; j < 4; ++j) {
      int n = (wave * 4 + j) * 16 + col;
#pragma unroll
      for (int v = 0; v < 8; ++v) {
        long o = (long)(m0 + mt * 16 + rowoff + v) * DH + n;
        float x = H0[o] + acc[mt][j][v];
        Hn[o] = x > 0.0f ? x : 0.0f;
      }
    }
  }
}

// ----- GEMM 3: H_v = relu([V ; M_v] @ W_o + b_o), K = 672 -------------------
__global__ __launch_bounds__(256) void gemm_output_kernel(
    const float* __restrict__ V, const float* __restrict__ Mv,
    const unsigned short* __restrict__ Wp, const float* __restrict__ b_o,
    float* __restrict__ Hv)
{
  __shared__ unsigned short Apack[2 * KTO * 32 * 16];   // 42 KB
  const int m0 = blockIdx.x * 32;
  const int t  = threadIdx.x;

  for (int idx = t; idx < 32 * KO; idx += 256) {
    int r = idx / KO, k = idx - r * KO;
    int a = m0 + r;
    float v;
    if (k < D_V)            v = V[(long)a * D_V + k];
    else if (k < D_V + DH)  v = Mv[(long)a * DH + (k - D_V)];
    else                    v = 0.0f;
    int mt = r >> 4, rr = r & 15;
    int kt = k >> 5, kk = k & 31;
    int lane = rr + ((kk & 8) << 1);
    int h    = (kk & 7) | ((kk & 16) >> 1);
    Apack[((mt * KTO + kt) * 32 + lane) * 16 + h] = f2bf(v);
  }
  __syncthreads();

  const int lane = t & 31, wave = t >> 5;
  v8f acc[2][4] = {};
  for (int kt = 0; kt < KTO; ++kt) {
    v16bf a0 = __builtin_bit_cast(v16bf, *(const v8u*)&Apack[(kt * 32 + lane) * 16]);
    v16bf a1 = __builtin_bit_cast(v16bf, *(const v8u*)&Apack[(((KTO + kt) * 32) + lane) * 16]);
#pragma unroll
    for (int j = 0; j < 4; ++j) {
      int nt = wave * 4 + j;
      v16bf b = __builtin_bit_cast(v16bf,
          *(const v8u*)&Wp[((long)(kt * NT_TOTAL + nt) * 32 + lane) * 16]);
      acc[0][j] = __builtin_amdgcn_wmma_f32_16x16x32_bf16(
          false, a0, false, b, (short)0, acc[0][j], false, false);
      acc[1][j] = __builtin_amdgcn_wmma_f32_16x16x32_bf16(
          false, a1, false, b, (short)0, acc[1][j], false, false);
    }
  }
  const int col = lane & 15, rowoff = (lane >> 4) * 8;
#pragma unroll
  for (int mt = 0; mt < 2; ++mt) {
#pragma unroll
    for (int j = 0; j < 4; ++j) {
      int n = (wave * 4 + j) * 16 + col;
      float bias = b_o[n];
#pragma unroll
      for (int v = 0; v < 8; ++v) {
        long o = (long)(m0 + mt * 16 + rowoff + v) * DH + n;
        float x = acc[mt][j][v] + bias;
        Hv[o] = x > 0.0f ? x : 0.0f;
      }
    }
  }
}

// ---------------- segment-sum of edge states into nodes ----------------------
__global__ __launch_bounds__(256) void scatter_kernel(
    const float* __restrict__ Hc, const int* __restrict__ dst,
    float* __restrict__ Mn)
{
  int e = blockIdx.x;
  int d = dst[e];
  const float* hr = Hc + (long)e * DH;
  float* mr = Mn + (long)d * DH;
  for (int k = threadIdx.x; k < DH; k += 256) atomicAdd(&mr[k], hr[k]);
}

// ---------------- molecule aggregation ---------------------------------------
__global__ __launch_bounds__(256) void count_kernel(
    const int* __restrict__ batch, float* __restrict__ counts)
{
  int a = blockIdx.x * 256 + threadIdx.x;
  if (a < N_ATOMS) atomicAdd(&counts[batch[a]], 1.0f);
}

__global__ __launch_bounds__(256) void aggregate_kernel(
    const float* __restrict__ Hv, const int* __restrict__ batch,
    float* __restrict__ sums)
{
  long idx = (long)blockIdx.x * 256 + threadIdx.x;
  if (idx >= (long)N_ATOMS * DH) return;
  int a = (int)(idx >> 9), k = (int)(idx & 511);
  atomicAdd(&sums[(long)batch[a] * DH + k], Hv[idx]);
}

__global__ __launch_bounds__(256) void mean_kernel(
    float* __restrict__ sums, const float* __restrict__ counts)
{
  int idx = blockIdx.x * 256 + threadIdx.x;     // N_MOLS*DH = 2M
  int m = idx >> 9;
  sums[idx] = sums[idx] / fmaxf(counts[m], 1.0f);
}

// ---------------- batchnorm (training-mode batch stats, biased var) ----------
__global__ __launch_bounds__(256) void bn_stats_kernel(
    const float* __restrict__ hm, float* __restrict__ mu, float* __restrict__ var)
{
  int c = blockIdx.x;
  int t = threadIdx.x;
  float s = 0.0f, s2 = 0.0f;
  for (int m = t; m < N_MOLS; m += 256) {
    float x = hm[(long)m * DH + c];
    s += x; s2 += x * x;
  }
  __shared__ float rs[256], rs2[256];
  rs[t] = s; rs2[t] = s2;
  __syncthreads();
  for (int o = 128; o > 0; o >>= 1) {
    if (t < o) { rs[t] += rs[t + o]; rs2[t] += rs2[t + o]; }
    __syncthreads();
  }
  if (t == 0) {
    float m_ = rs[0] / (float)N_MOLS;
    mu[c]  = m_;
    var[c] = rs2[0] / (float)N_MOLS - m_ * m_;
  }
}

__global__ __launch_bounds__(256) void bn_norm_kernel(
    const float* __restrict__ hm, const float* __restrict__ mu,
    const float* __restrict__ var, const float* __restrict__ gamma,
    const float* __restrict__ beta, float* __restrict__ out)
{
  int idx = blockIdx.x * 256 + threadIdx.x;     // N_MOLS*DH
  int c = idx & 511;
  out[idx] = (hm[idx] - mu[c]) * rsqrtf(var[c] + BN_EPS) * gamma[c] + beta[c];
}

// -----------------------------------------------------------------------------
extern "C" void kernel_launch(void* const* d_in, const int* in_sizes, int n_in,
                              void* d_out, int out_size, void* d_ws, size_t ws_size,
                              hipStream_t stream) {
  (void)in_sizes; (void)n_in; (void)out_size; (void)ws_size;
  const float* V        = (const float*)d_in[0];
  const float* E        = (const float*)d_in[1];
  const int*   edge_src = (const int*)  d_in[2];
  const int*   edge_dst = (const int*)  d_in[3];
  const int*   batch    = (const int*)  d_in[4];
  const float* W_i      = (const float*)d_in[5];
  const float* W_h      = (const float*)d_in[6];
  const float* W_o      = (const float*)d_in[7];
  const float* b_o      = (const float*)d_in[8];
  const float* gamma    = (const float*)d_in[9];
  const float* beta     = (const float*)d_in[10];
  float* out = (float*)d_out;

  // -------- workspace carve-up (256B aligned) --------
  char* base = (char*)d_ws;
  size_t off = 0;
  auto take = [&](size_t bytes) -> char* {
    char* p = base + off;
    off = (off + bytes + 255) & ~(size_t)255;
    return p;
  };
  unsigned short* Wi_p = (unsigned short*)take((size_t)KTI * 16384 * 2);
  unsigned short* Wh_p = (unsigned short*)take((size_t)KTH * 16384 * 2);
  unsigned short* Wo_p = (unsigned short*)take((size_t)KTO * 16384 * 2);
  float* H0   = (float*)take((size_t)N_EDGES * DH * 4);
  float* Ha   = (float*)take((size_t)N_EDGES * DH * 4);
  float* Hb   = (float*)take((size_t)N_EDGES * DH * 4);
  float* Mn   = (float*)take((size_t)N_ATOMS * DH * 4);
  float* sums = (float*)take((size_t)N_MOLS * DH * 4);
  float* counts = (float*)take((size_t)N_MOLS * 4);
  float* mu   = (float*)take(DH * 4);
  float* var  = (float*)take(DH * 4);

  // -------- weight repack to WMMA bf16 layout --------
  pack_weight_kernel<<<(KTI * 16384 + 255) / 256, 256, 0, stream>>>(W_i, Wi_p, D_V + D_E, KTI);
  pack_weight_kernel<<<(KTH * 16384 + 255) / 256, 256, 0, stream>>>(W_h, Wh_p, DH, KTH);
  pack_weight_kernel<<<(KTO * 16384 + 255) / 256, 256, 0, stream>>>(W_o, Wo_p, D_V + DH, KTO);

  // -------- H0 = [V[src];E] @ W_i ; H = relu(H0) --------
  gemm_input_kernel<<<N_EDGES / 32, 256, 0, stream>>>(V, E, edge_src, Wi_p, H0, Ha);

  // -------- DEPTH-1 = 2 message passing iterations (ping-pong Ha/Hb) --------
  float* cur = Ha;
  float* nxt = Hb;
  for (int it = 0; it < 2; ++it) {
    hipMemsetAsync(Mn, 0, (size_t)N_ATOMS * DH * 4, stream);
    scatter_kernel<<<N_EDGES, 256, 0, stream>>>(cur, edge_dst, Mn);
    gemm_update_kernel<<<N_EDGES / 32, 256, 0, stream>>>(Mn, cur, edge_src, Wh_p, H0, nxt);
    float* tmp = cur; cur = nxt; nxt = tmp;
  }

  // -------- finalize: M_v = segsum(H, dst); H_v = relu([V;M_v]@W_o + b_o) ----
  hipMemsetAsync(Mn, 0, (size_t)N_ATOMS * DH * 4, stream);
  scatter_kernel<<<N_EDGES, 256, 0, stream>>>(cur, edge_dst, Mn);
  float* Hv = nxt;   // free ping-pong buffer reused for node states
  gemm_output_kernel<<<N_ATOMS / 32, 256, 0, stream>>>(V, Mn, Wo_p, b_o, Hv);

  // -------- mean aggregation over molecules --------
  hipMemsetAsync(sums, 0, (size_t)N_MOLS * DH * 4, stream);
  hipMemsetAsync(counts, 0, (size_t)N_MOLS * 4, stream);
  count_kernel<<<(N_ATOMS + 255) / 256, 256, 0, stream>>>(batch, counts);
  aggregate_kernel<<<(int)(((long)N_ATOMS * DH + 255) / 256), 256, 0, stream>>>(Hv, batch, sums);
  mean_kernel<<<(N_MOLS * DH) / 256, 256, 0, stream>>>(sums, counts);

  // -------- batchnorm --------
  bn_stats_kernel<<<DH, 256, 0, stream>>>(sums, mu, var);
  bn_norm_kernel<<<(N_MOLS * DH) / 256, 256, 0, stream>>>(sums, mu, var, gamma, beta, out);
}
